// RefGatedMLPFusedMoE_47562467836577
// MI455X (gfx1250) — matrix operand order repaired and checked by reading, not compiled, not run on hardware
//
#include <hip/hip_runtime.h>
#include <hip/hip_bf16.h>

// ---------------------------------------------------------------------------
// MoE gated MLP (top-2, renormalized) for MI455X / gfx1250 (wave32, WMMA).
// T=4096, H=1024, I=4096, E=8.
// route -> gather per expert, bf16 WMMA 16x16x32 GEMMs (f32 accum),
// fused SiLU, f32 atomic scatter-add for the down projection.
// LDS tiles padded to 40 elems/row (80B = 20 dwords) -> conflict-free banking.
// A tiles staged with global_load_async_to_lds_b128 when available.
// ---------------------------------------------------------------------------

typedef __bf16 bf16;
typedef __attribute__((ext_vector_type(16))) __bf16 v16bf;
typedef __attribute__((ext_vector_type(8)))  __bf16 v8bf;
typedef __attribute__((ext_vector_type(8)))  float  v8f;
typedef __attribute__((ext_vector_type(4)))  int    v4i;

typedef union { v16bf v; v8bf h[2]; } FragBF;

#define T_TOK 4096
#define HDIM  1024
#define IDIM  4096
#define NEXP  8
#define NPAIR (T_TOK * 2)
#define LPAD  40                 // padded LDS row length (elements)

#if __has_builtin(__builtin_amdgcn_global_load_async_to_lds_b128)
#define HAVE_ASYNC_LDS 1
#else
#define HAVE_ASYNC_LDS 0
#endif

#if HAVE_ASYNC_LDS
__device__ __forceinline__ void async_cp16(const bf16* g, bf16* l) {
    __builtin_amdgcn_global_load_async_to_lds_b128(
        (__attribute__((address_space(1))) v4i*)(g),
        (__attribute__((address_space(3))) v4i*)(l), 0, 0);
}
__device__ __forceinline__ void async_wait0() {
#if __has_builtin(__builtin_amdgcn_s_wait_asynccnt)
    __builtin_amdgcn_s_wait_asynccnt(0);
#else
    asm volatile("s_wait_asynccnt 0x0" ::: "memory");
#endif
}
#endif

// workspace layout (bytes)
static constexpr size_t XBF_OFF  = 0;                                   // X bf16: 8 MB
static constexpr size_t TOK_OFF  = XBF_OFF  + (size_t)T_TOK * HDIM * 2; // int[NPAIR]
static constexpr size_t WMAP_OFF = TOK_OFF  + (size_t)NPAIR * 4;        // float[NPAIR]
static constexpr size_t TOP_OFF  = WMAP_OFF + (size_t)NPAIR * 4;        // int[NPAIR]
static constexpr size_t TW_OFF   = TOP_OFF  + (size_t)NPAIR * 4;        // float[NPAIR]
static constexpr size_t CNT_OFF  = TW_OFF   + (size_t)NPAIR * 4;        // counts/bases/cursors
static constexpr size_t ACT_OFF  = CNT_OFF  + 256;                      // bf16[NPAIR*IDIM] = 64 MB

// ------------------------------- small kernels ------------------------------

__global__ void moe_init_small(int* cnt) {
    int t = threadIdx.x;
    if (t < 24) cnt[t] = 0;
}

__global__ void moe_zero_out(float4* out, int n4) {
    int i = blockIdx.x * 256 + threadIdx.x;
    if (i < n4) out[i] = make_float4(0.f, 0.f, 0.f, 0.f);
}

__global__ void moe_cvt_x(const float* __restrict__ X, bf16* __restrict__ Xbf, int n8) {
    int i = blockIdx.x * 256 + threadIdx.x;
    if (i >= n8) return;
    const float4* src = (const float4*)X + (size_t)i * 2;
    float4 a = src[0], b = src[1];
    union { bf16 e[8]; uint4 u; } p;
    p.e[0] = (bf16)a.x; p.e[1] = (bf16)a.y; p.e[2] = (bf16)a.z; p.e[3] = (bf16)a.w;
    p.e[4] = (bf16)b.x; p.e[5] = (bf16)b.y; p.e[6] = (bf16)b.z; p.e[7] = (bf16)b.w;
    *(uint4*)(Xbf + (size_t)i * 8) = p.u;
}

__global__ void moe_router_topk(const float* __restrict__ logits,
                                int* __restrict__ topidx, float* __restrict__ topw,
                                int* __restrict__ cnt) {
    int t = blockIdx.x * 256 + threadIdx.x;
    if (t >= T_TOK) return;
    float best = -1e30f, second = -1e30f;
    int bi = 0, si = 0;
    #pragma unroll
    for (int e = 0; e < NEXP; ++e) {
        float v = logits[t * NEXP + e];
        if (v > best)        { second = best; si = bi; best = v; bi = e; }
        else if (v > second) { second = v; si = e; }
    }
    float ex = __expf(second - best);
    float w0 = 1.f / (1.f + ex);
    float w1 = ex * w0;
    topidx[t * 2 + 0] = bi;  topw[t * 2 + 0] = w0;
    topidx[t * 2 + 1] = si;  topw[t * 2 + 1] = w1;
    atomicAdd(&cnt[bi], 1);
    atomicAdd(&cnt[si], 1);
}

__global__ void moe_router_prefix(int* cnt) {
    if (threadIdx.x == 0) {
        int s = 0;
        for (int e = 0; e < NEXP; ++e) { cnt[8 + e] = s; s += cnt[e]; }
    }
}

__global__ void moe_router_scatter(const int* __restrict__ topidx, const float* __restrict__ topw,
                                   int* __restrict__ cnt,
                                   int* __restrict__ tokmap, float* __restrict__ wmap) {
    int t = blockIdx.x * 256 + threadIdx.x;
    if (t >= T_TOK) return;
    #pragma unroll
    for (int k = 0; k < 2; ++k) {
        int e = topidx[t * 2 + k];
        int pos = atomicAdd(&cnt[16 + e], 1);
        int row = cnt[8 + e] + pos;
        tokmap[row] = t;
        wmap[row] = topw[t * 2 + k];
    }
}

// stage one [32][128] f32 weight chunk as bf16 into [n][k]-padded LDS,
// packing k-quads so stores are 8-byte and conversions pair into cvt_pk.
__device__ __forceinline__ void stage_w(const float* __restrict__ wp, size_t ldw,
                                        int kb, int tid, bf16 (*B)[LPAD]) {
    const int kq   = (tid >> 5) * 4;       // k: 0,4,...,28 (one quad per wave)
    const int ncol = (tid & 31) * 4;       // n: lane*4
    float4 r0 = *(const float4*)(wp + (size_t)(kb + kq + 0) * ldw + ncol);
    float4 r1 = *(const float4*)(wp + (size_t)(kb + kq + 1) * ldw + ncol);
    float4 r2 = *(const float4*)(wp + (size_t)(kb + kq + 2) * ldw + ncol);
    float4 r3 = *(const float4*)(wp + (size_t)(kb + kq + 3) * ldw + ncol);
    union { bf16 e[4]; uint2 u; } pk;
    pk.e[0] = (bf16)r0.x; pk.e[1] = (bf16)r1.x; pk.e[2] = (bf16)r2.x; pk.e[3] = (bf16)r3.x;
    *(uint2*)(&B[ncol + 0][kq]) = pk.u;
    pk.e[0] = (bf16)r0.y; pk.e[1] = (bf16)r1.y; pk.e[2] = (bf16)r2.y; pk.e[3] = (bf16)r3.y;
    *(uint2*)(&B[ncol + 1][kq]) = pk.u;
    pk.e[0] = (bf16)r0.z; pk.e[1] = (bf16)r1.z; pk.e[2] = (bf16)r2.z; pk.e[3] = (bf16)r3.z;
    *(uint2*)(&B[ncol + 2][kq]) = pk.u;
    pk.e[0] = (bf16)r0.w; pk.e[1] = (bf16)r1.w; pk.e[2] = (bf16)r2.w; pk.e[3] = (bf16)r3.w;
    *(uint2*)(&B[ncol + 3][kq]) = pk.u;
}

// ------------------------------- GEMM 1 -------------------------------------
// act[row, i] = silu(X[tok] @ Wg[e])[i] * (X[tok] @ Wu[e])[i]
// block tile 128(M) x 128(N), 8 waves in 4(M) x 2(N), K-step 32 (bf16 WMMA).
__global__ void __launch_bounds__(256)
moe_gemm1(const float* __restrict__ Wg, const float* __restrict__ Wu,
          const bf16* __restrict__ Xbf,
          const int* __restrict__ tokmap, const int* __restrict__ counts,
          const int* __restrict__ bases, bf16* __restrict__ act) {
    const int e     = blockIdx.z;
    const int cnt   = counts[e];
    const int mtile = blockIdx.y * 128;
    if (mtile >= cnt) return;
    const int base = bases[e];
    const int iblk = blockIdx.x * 128;
    const int tid  = threadIdx.x;
    const int lane = tid & 31;
    const int wid  = tid >> 5;
    const int wm   = wid & 3;    // rows [wm*32, +32)
    const int wn   = wid >> 2;   // cols [wn*64, +64)

    __shared__ bf16 Al[128][LPAD];   // [m][k]
    __shared__ bf16 Bg[128][LPAD];   // [n][k]
    __shared__ bf16 Bu[128][LPAD];

    const int arow  = tid >> 1;
    const int akoff = (tid & 1) * 16;
    int tok = 0;
    if (mtile + arow < cnt) tok = tokmap[base + mtile + arow];
    const bf16* xrow = Xbf + (size_t)tok * HDIM;

    const size_t wbase = (size_t)e * HDIM * IDIM;
    const float* wgp = Wg + wbase + iblk;
    const float* wup = Wu + wbase + iblk;

    v8f zacc = {};
    v8f ag[2][4], au[2][4];
    #pragma unroll
    for (int i = 0; i < 2; ++i)
        #pragma unroll
        for (int j = 0; j < 4; ++j) { ag[i][j] = zacc; au[i][j] = zacc; }

    const int half = lane >> 4;
    const int l16  = lane & 15;

    for (int kb = 0; kb < HDIM; kb += 32) {
        __syncthreads();
        // A tile: gathered bf16 token rows, 32 B/thread
#if HAVE_ASYNC_LDS
        async_cp16(xrow + kb + akoff,     &Al[arow][akoff]);
        async_cp16(xrow + kb + akoff + 8, &Al[arow][akoff + 8]);
#else
        {
            const uint4* src = (const uint4*)(xrow + kb + akoff);
            *(uint4*)(&Al[arow][akoff])     = src[0];
            *(uint4*)(&Al[arow][akoff + 8]) = src[1];
        }
#endif
        // B tiles: f32 -> bf16, k-quad packed, [n][k] layout
        stage_w(wgp, IDIM, kb, tid, Bg);
        stage_w(wup, IDIM, kb, tid, Bu);
        if (kb + 32 < HDIM) {
            __builtin_prefetch(wgp + (size_t)(kb + 32 + (tid >> 5) * 4) * IDIM + (tid & 31) * 4, 0, 0);
            __builtin_prefetch(wup + (size_t)(kb + 32 + (tid >> 5) * 4) * IDIM + (tid & 31) * 4, 0, 0);
        }
#if HAVE_ASYNC_LDS
        async_wait0();
#endif
        __syncthreads();

        FragBF afr[2], bgfr[4], bufr[4];
        #pragma unroll
        for (int sm = 0; sm < 2; ++sm) {
            int r = wm * 32 + sm * 16 + l16;
            afr[sm].h[0] = *(const v8bf*)(&Al[r][half * 16]);
            afr[sm].h[1] = *(const v8bf*)(&Al[r][half * 16 + 8]);
        }
        #pragma unroll
        for (int sn = 0; sn < 4; ++sn) {
            int n = wn * 64 + sn * 16 + l16;
            bgfr[sn].h[0] = *(const v8bf*)(&Bg[n][half * 16]);
            bgfr[sn].h[1] = *(const v8bf*)(&Bg[n][half * 16 + 8]);
            bufr[sn].h[0] = *(const v8bf*)(&Bu[n][half * 16]);
            bufr[sn].h[1] = *(const v8bf*)(&Bu[n][half * 16 + 8]);
        }
        #pragma unroll
        for (int sm = 0; sm < 2; ++sm)
            #pragma unroll
            for (int sn = 0; sn < 4; ++sn) {
                ag[sm][sn] = __builtin_amdgcn_wmma_f32_16x16x32_bf16(
                    false, afr[sm].v, false, bgfr[sn].v, (short)0, ag[sm][sn], false, false);
                au[sm][sn] = __builtin_amdgcn_wmma_f32_16x16x32_bf16(
                    false, afr[sm].v, false, bufr[sn].v, (short)0, au[sm][sn], false, false);
            }
    }

    // fused SiLU(gate) * up -> bf16 activation buffer
    #pragma unroll
    for (int sm = 0; sm < 2; ++sm)
        #pragma unroll
        for (int sn = 0; sn < 4; ++sn)
            #pragma unroll
            for (int q = 0; q < 8; ++q) {
                int m = wm * 32 + sm * 16 + half * 8 + q;
                int n = wn * 64 + sn * 16 + l16;
                if (mtile + m < cnt) {
                    float g = ag[sm][sn][q];
                    float u = au[sm][sn][q];
                    float s = g / (1.f + __expf(-g));
                    act[(size_t)(base + mtile + m) * IDIM + iblk + n] = (bf16)(s * u);
                }
            }
}

// ------------------------------- GEMM 2 -------------------------------------
// out[tok] += w * (act[row] @ Wd[e]);  K = IDIM, same tiling.
__global__ void __launch_bounds__(256)
moe_gemm2(const float* __restrict__ Wd, const bf16* __restrict__ act,
          const int* __restrict__ tokmap, const float* __restrict__ wmap,
          const int* __restrict__ counts, const int* __restrict__ bases,
          float* __restrict__ out) {
    const int e     = blockIdx.z;
    const int cnt   = counts[e];
    const int mtile = blockIdx.y * 128;
    if (mtile >= cnt) return;
    const int base = bases[e];
    const int hblk = blockIdx.x * 128;
    const int tid  = threadIdx.x;
    const int lane = tid & 31;
    const int wid  = tid >> 5;
    const int wm   = wid & 3;
    const int wn   = wid >> 2;

    __shared__ bf16 Al[128][LPAD];
    __shared__ bf16 Bd[128][LPAD];

    const int arow  = tid >> 1;
    const int akoff = (tid & 1) * 16;
    int gr = base + mtile + arow;
    if (gr > NPAIR - 1) gr = NPAIR - 1;
    const bf16* arp = act + (size_t)gr * IDIM;

    const float* wdp = Wd + (size_t)e * IDIM * HDIM + hblk;

    v8f zacc = {};
    v8f acc[2][4];
    #pragma unroll
    for (int i = 0; i < 2; ++i)
        #pragma unroll
        for (int j = 0; j < 4; ++j) acc[i][j] = zacc;

    const int half = lane >> 4;
    const int l16  = lane & 15;

    for (int kb = 0; kb < IDIM; kb += 32) {
        __syncthreads();
#if HAVE_ASYNC_LDS
        async_cp16(arp + kb + akoff,     &Al[arow][akoff]);
        async_cp16(arp + kb + akoff + 8, &Al[arow][akoff + 8]);
#else
        {
            const uint4* src = (const uint4*)(arp + kb + akoff);
            *(uint4*)(&Al[arow][akoff])     = src[0];
            *(uint4*)(&Al[arow][akoff + 8]) = src[1];
        }
#endif
        stage_w(wdp, HDIM, kb, tid, Bd);
        if (kb + 32 < IDIM)
            __builtin_prefetch(wdp + (size_t)(kb + 32 + (tid >> 5) * 4) * HDIM + (tid & 31) * 4, 0, 0);
#if HAVE_ASYNC_LDS
        async_wait0();
#endif
        __syncthreads();

        FragBF afr[2], bfr[4];
        #pragma unroll
        for (int sm = 0; sm < 2; ++sm) {
            int r = wm * 32 + sm * 16 + l16;
            afr[sm].h[0] = *(const v8bf*)(&Al[r][half * 16]);
            afr[sm].h[1] = *(const v8bf*)(&Al[r][half * 16 + 8]);
        }
        #pragma unroll
        for (int sn = 0; sn < 4; ++sn) {
            int n = wn * 64 + sn * 16 + l16;
            bfr[sn].h[0] = *(const v8bf*)(&Bd[n][half * 16]);
            bfr[sn].h[1] = *(const v8bf*)(&Bd[n][half * 16 + 8]);
        }
        #pragma unroll
        for (int sm = 0; sm < 2; ++sm)
            #pragma unroll
            for (int sn = 0; sn < 4; ++sn)
                acc[sm][sn] = __builtin_amdgcn_wmma_f32_16x16x32_bf16(
                    false, afr[sm].v, false, bfr[sn].v, (short)0, acc[sm][sn], false, false);
    }

    #pragma unroll
    for (int sm = 0; sm < 2; ++sm)
        #pragma unroll
        for (int sn = 0; sn < 4; ++sn)
            #pragma unroll
            for (int q = 0; q < 8; ++q) {
                int m = wm * 32 + sm * 16 + half * 8 + q;
                int n = wn * 64 + sn * 16 + l16;
                if (mtile + m < cnt) {
                    int row = base + mtile + m;
                    int tok = tokmap[row];
                    float w = wmap[row];
                    atomicAdd(&out[(size_t)tok * HDIM + hblk + n], w * acc[sm][sn][q]);
                }
            }
}

// ------------------------------- launcher -----------------------------------

extern "C" void kernel_launch(void* const* d_in, const int* in_sizes, int n_in,
                              void* d_out, int out_size, void* d_ws, size_t ws_size,
                              hipStream_t stream) {
    const float* X  = (const float*)d_in[0];   // [T,H]
    const float* RL = (const float*)d_in[1];   // [T,E]
    const float* Wg = (const float*)d_in[2];   // [E,H,I]
    const float* Wu = (const float*)d_in[3];   // [E,H,I]
    const float* Wd = (const float*)d_in[4];   // [E,I,H]
    float* out = (float*)d_out;

    char* ws = (char*)d_ws;
    bf16*  Xbf    = (bf16*)(ws + XBF_OFF);
    int*   tokmap = (int*)(ws + TOK_OFF);
    float* wmap   = (float*)(ws + WMAP_OFF);
    int*   topidx = (int*)(ws + TOP_OFF);
    float* topw   = (float*)(ws + TW_OFF);
    int*   cnt    = (int*)(ws + CNT_OFF);
    bf16*  act    = (bf16*)(ws + ACT_OFF);

    moe_init_small<<<1, 32, 0, stream>>>(cnt);
    moe_zero_out<<<(T_TOK * HDIM / 4 + 255) / 256, 256, 0, stream>>>((float4*)out, T_TOK * HDIM / 4);
    moe_cvt_x<<<(T_TOK * HDIM / 8 + 255) / 256, 256, 0, stream>>>(X, Xbf, T_TOK * HDIM / 8);
    moe_router_topk<<<T_TOK / 256, 256, 0, stream>>>(RL, topidx, topw, cnt);
    moe_router_prefix<<<1, 32, 0, stream>>>(cnt);
    moe_router_scatter<<<T_TOK / 256, 256, 0, stream>>>(topidx, topw, cnt, tokmap, wmap);

    dim3 g1(IDIM / 128, T_TOK / 128, NEXP);   // tiles past count[e] early-exit
    moe_gemm1<<<g1, 256, 0, stream>>>(Wg, Wu, Xbf, tokmap, cnt, cnt + 8, act);

    dim3 g2(HDIM / 128, T_TOK / 128, NEXP);
    moe_gemm2<<<g2, 256, 0, stream>>>(Wd, act, tokmap, wmap, cnt, cnt + 8, out);
}
